// KEDGN_7249904795896
// MI455X (gfx1250) — matrix operand order repaired and checked by reading, not compiled, not run on hardware
//
#include <hip/hip_runtime.h>
#include <math.h>

#define BB 32
#define TT 96
#define NN 128
#define DM 64
#define QD 5
#define NE 8
#define PLMD 768
#define CC 129
#define DK 16
#define KPAD 160
#define NPADV 144
#define CPAD 132
#define NEGV (-9e15f)

typedef __attribute__((ext_vector_type(16))) __bf16 v16bf;
typedef __attribute__((ext_vector_type(8)))  float  v8f;
typedef __attribute__((ext_vector_type(4)))  unsigned int u32x4;

// ---- swizzled-offset helpers: map (row,col) -> tile-contiguous lane layout
__device__ __forceinline__ int swA(int m, int k, int nkt) {
  int mt = m >> 4, lm = m & 15;
  int kt = k >> 5, kk = k & 31;
  int hi = (kk >> 3) & 1;
  int h  = (kk & 7) + ((kk >> 4) & 1) * 8;
  return (((mt * nkt + kt) * 32) + hi * 16 + lm) * 16 + h;
}
__device__ __forceinline__ int swB(int k, int n, int nnt) {
  int kt = k >> 5, kk = k & 31;
  int hi = (kk >> 3) & 1, h = (kk & 7) + ((kk >> 4) & 1) * 8;
  int nt = n >> 4, ln = n & 15;
  return (((kt * nnt + nt) * 32) + hi * 16 + ln) * 16 + h;
}
// Load one lane's 16 bf16 operand halves: two aligned b128 loads.
__device__ __forceinline__ v16bf ldTile(const __bf16* base, int tileIdx, int lane) {
  const u32x4* p = (const u32x4*)(base + ((long long)tileIdx * 32 + lane) * 16);
  union { struct { u32x4 a, b; } w; v16bf v; } u;
  u.w.a = p[0];
  u.w.b = p[1];
  return u.v;
}
__device__ __forceinline__ v8f wmma_bf16(v16bf a, v16bf b, v8f c) {
  return __builtin_amdgcn_wmma_f32_16x16x32_bf16(false, a, false, b, (short)0, c, false, false);
}

// ---- one-time: qv [N,QD], normalized emb [N,NE] -------------------------
__global__ void k_precompute(const float* plm,
                             const float* pfW1, const float* pfb1,
                             const float* pfW2, const float* pfb2,
                             const float* pgW1, const float* pgb1,
                             const float* pgW2, const float* pgb2,
                             float* qv, float* embn) {
  __shared__ float s_hid[NN][128];
  int tid = threadIdx.x;
  for (int idx = tid; idx < NN * 128; idx += 256) {
    int n = idx >> 7, j = idx & 127;
    float acc = pfb1[j];
    for (int p = 0; p < PLMD; ++p) acc += plm[n * PLMD + p] * pfW1[p * 128 + j];
    s_hid[n][j] = fmaxf(acc, 0.f);
  }
  __syncthreads();
  for (int idx = tid; idx < NN * QD; idx += 256) {
    int n = idx / QD, o = idx % QD;
    float acc = pfb2[o];
    for (int j = 0; j < 128; ++j) acc += s_hid[n][j] * pfW2[j * QD + o];
    qv[n * QD + o] = acc;
  }
  __syncthreads();
  for (int idx = tid; idx < NN * 128; idx += 256) {
    int n = idx >> 7, j = idx & 127;
    float acc = pgb1[j];
    for (int p = 0; p < PLMD; ++p) acc += plm[n * PLMD + p] * pgW1[p * 128 + j];
    s_hid[n][j] = fmaxf(acc, 0.f);
  }
  __syncthreads();
  if (tid < NN) {
    int n = tid;
    float e[NE]; float ss = 0.f;
    for (int o = 0; o < NE; ++o) {
      float acc = pgb2[o];
      for (int j = 0; j < 128; ++j) acc += s_hid[n][j] * pgW2[j * NE + o];
      e[o] = acc; ss += acc * acc;
    }
    float inv = 1.f / fmaxf(sqrtf(ss), 1e-12f);
    for (int o = 0; o < NE; ++o) embn[n * NE + o] = e[o] * inv;
  }
}

__global__ void k_adj(const float* embn, float* adj) {
  __shared__ float s_e[NN][NE];
  int tid = threadIdx.x;
  for (int idx = tid; idx < NN * NE; idx += 128) s_e[idx / NE][idx % NE] = embn[idx];
  __syncthreads();
  int r = tid;
  float mx = -1e30f;
  for (int m = 0; m < NN; ++m) {
    float d = 0.f;
    for (int o = 0; o < NE; ++o) d += s_e[r][o] * s_e[m][o];
    mx = fmaxf(mx, d);
  }
  float sum = 0.f;
  for (int m = 0; m < NN; ++m) {
    float d = 0.f;
    for (int o = 0; o < NE; ++o) d += s_e[r][o] * s_e[m][o];
    sum += expf(d - mx);
  }
  float inv = 1.f / sum;
  for (int m = 0; m < NN; ++m) {
    float d = 0.f;
    for (int o = 0; o < NE; ++o) d += s_e[r][o] * s_e[m][o];
    adj[r * NN + m] = expf(d - mx) * inv;
  }
}

__global__ void k_vartotal(const int* mask, float* vt) {
  int idx = blockIdx.x * 256 + threadIdx.x;
  if (idx >= BB * NN) return;
  int b = idx >> 7, n = idx & 127;
  float s = 0.f;
  for (int t = 0; t < TT; ++t) s += (float)mask[(b * TT + t) * NN + n];
  vt[idx] = s;
}

// ---- one-time: hypernet weights Wn + Wv, stored B-swizzled bf16 ---------
__global__ void k_genw(const float* qv,
                       const float* gW0, const float* gb0,
                       const float* gW1, const float* gb1,
                       const float* gW2, const float* gb2,
                       const float* Wv,
                       __bf16* Wn, float* bn, __bf16* Wvb) {
  long long idx = (long long)blockIdx.x * 256 + threadIdx.x;
  const long long NWN = 3LL * NN * KPAD * DM;
  if (idx < NWN) {
    int o = (int)(idx % DM); long long r = idx / DM;
    int i = (int)(r % KPAD); r /= KPAD;
    int n = (int)(r % NN);   int g = (int)(r / NN);
    float val = 0.f;
    if (i < CC) {
      const float* W = (g == 0) ? gW0 : (g == 1) ? gW1 : gW2;
      for (int d = 0; d < QD; ++d) val += qv[n * QD + d] * W[(d * CC + i) * DM + o];
    }
    Wn[(long long)(g * NN + n) * KPAD * DM + swB(i, o, DM / 16)] = (__bf16)val;
    return;
  }
  idx -= NWN;
  if (idx < 3 * NN * DM) {
    int o = (int)(idx % DM); int r = (int)(idx / DM);
    int n = r % NN; int g = r / NN;
    const float* bs = (g == 0) ? gb0 : (g == 1) ? gb1 : gb2;
    float val = 0.f;
    for (int d = 0; d < QD; ++d) val += qv[n * QD + d] * bs[d * DM + o];
    bn[(g * NN + n) * DM + o] = val;
    return;
  }
  idx -= 3 * NN * DM;
  if (idx < KPAD * NPADV) {
    int o = (int)(idx % NPADV); int i = (int)(idx / NPADV);
    float val = (i < CC && o < CC) ? Wv[i * CC + o] : 0.f;
    Wvb[swB(i, o, NPADV / 16)] = (__bf16)val;
  }
}

// ---- per step: comb (A-swizzled LDS), q/k, v = comb@Wv (WMMA) -----------
__global__ void k_qkv(int t, const float* obs, const float* ai, const float* vt,
                      const float* h, const float* Wq, const float* bq,
                      const float* Wk, const float* bk, const float* bv,
                      const __bf16* Wvb, float* qout, float* kout, __bf16* vout) {
  __shared__ __attribute__((aligned(16))) __bf16 s_cb[NN * KPAD]; // 40 KB, A-swizzled
  __shared__ float s_rsc[NN];
  int b = blockIdx.x, tid = threadIdx.x;
  if (tid < NN) {
    float a = ai[(b * TT + t) * NN + tid];
    s_rsc[tid] = 0.5f * tanhf(a / (vt[b * NN + tid] + 1.f));
  }
  __syncthreads();
  for (int idx = tid; idx < NN * KPAD; idx += 256) {
    int n = idx / KPAD, c = idx % KPAD;
    float v;
    if (c < DM)       v = obs[((b * TT + t) * NN + n) * DM + c];
    else if (c == DM) v = s_rsc[n];
    else if (c < CC)  v = h[(b * NN + n) * DM + (c - DM - 1)];
    else              v = 0.f;
    s_cb[swA(n, c, KPAD / 32)] = (__bf16)v;
  }
  __syncthreads();
  // q, k: plain dots through the swizzle map (small)
  for (int idx = tid; idx < NN * DK * 2; idx += 256) {
    int which = idx >= NN * DK;
    int l = which ? idx - NN * DK : idx;
    int n = l / DK, o = l % DK;
    const float* W = which ? Wk : Wq;
    float acc = which ? bk[o] : bq[o];
    for (int i = 0; i < CC; ++i) acc += (float)s_cb[swA(n, i, KPAD / 32)] * W[i * DK + o];
    (which ? kout : qout)[(b * NN + n) * DK + o] = acc;
  }
  // v = comb @ Wv + bv  (WMMA, M=128, N=144pad, K=160pad); output B-swizzled
  int lane = tid & 31, w = tid >> 5;
  int m0 = w * 16;
  int hi = (lane >> 4) & 1;
  __bf16* vo = vout + (long long)b * NN * NPADV;
  // hoist this wave's 5 A tiles into registers once (40 VGPRs)
  v16bf Ar[5];
#pragma unroll
  for (int kt = 0; kt < 5; ++kt) Ar[kt] = ldTile(s_cb, w * 5 + kt, lane);
#pragma unroll 1
  for (int nt = 0; nt < 9; ++nt) {
    int n0 = nt * 16;
    int ncol = n0 + (lane & 15);
    float bias = (ncol < CC) ? bv[ncol] : 0.f;
    v8f acc;
#pragma unroll
    for (int j = 0; j < 8; ++j) acc[j] = bias;
#pragma unroll
    for (int kt = 0; kt < 5; ++kt)
      acc = wmma_bf16(Ar[kt], ldTile(Wvb, kt * 9 + nt, lane), acc);
#pragma unroll
    for (int j = 0; j < 8; ++j) {
      int m = m0 + j + hi * 8;
      vo[swB(m, ncol, 9)] = (__bf16)acc[j];  // B-layout for att@v GEMM
    }
  }
}

// ---- per step: scores + softmax (A-swizzled LDS) + catt = att@v (WMMA) --
__global__ void k_att(int t, const float* ai, const float* vt, const int* mask,
                      const float* rw, const float* adj,
                      const float* qin, const float* kin,
                      const __bf16* vin, float* catt) {
  __shared__ float  s_q[NN][DK], s_k[NN][DK];                       // 16 KB
  __shared__ __attribute__((aligned(16))) __bf16 s_att[NN * NN];    // 32 KB, A-swizzled
  __shared__ float  s_rsc[NN], s_m[NN];
  int b = blockIdx.x, tid = threadIdx.x;
  if (tid < NN) {
    float a = ai[(b * TT + t) * NN + tid];
    s_rsc[tid] = 0.5f * tanhf(a / (vt[b * NN + tid] + 1.f));
    s_m[tid]   = (float)mask[(b * TT + t) * NN + tid];
  }
  for (int idx = tid; idx < NN * DK; idx += 256) {
    s_q[idx / DK][idx % DK] = qin[b * NN * DK + idx];
    s_k[idx / DK][idx % DK] = kin[b * NN * DK + idx];
  }
  __syncthreads();
  if (tid < NN) {
    int r = tid;
    float rr = s_rsc[r], mr = s_m[r];
    float mx = -1e30f, sum = 0.f;
    for (int m = 0; m < NN; ++m) {
      float d = 0.f;
      for (int o = 0; o < DK; ++o) d += s_q[r][o] * s_k[m][o];
      d *= 0.25f;  d = (d < 0.f) ? 0.2f * d : d;
      if (m != r) {
        float ca = adj[r * NN + m] *
                   (1.f - rw[r * NN + m] * fabsf(rr - s_rsc[m])) * (mr * s_m[m]);
        if (ca == 0.f) d = NEGV;
      }
      mx = fmaxf(mx, d);
    }
    for (int m = 0; m < NN; ++m) {
      float d = 0.f;
      for (int o = 0; o < DK; ++o) d += s_q[r][o] * s_k[m][o];
      d *= 0.25f;  d = (d < 0.f) ? 0.2f * d : d;
      if (m != r) {
        float ca = adj[r * NN + m] *
                   (1.f - rw[r * NN + m] * fabsf(rr - s_rsc[m])) * (mr * s_m[m]);
        if (ca == 0.f) d = NEGV;
      }
      sum += expf(d - mx);
    }
    float inv = 1.f / sum;
    for (int m = 0; m < NN; ++m) {
      float d = 0.f;
      for (int o = 0; o < DK; ++o) d += s_q[r][o] * s_k[m][o];
      d *= 0.25f;  d = (d < 0.f) ? 0.2f * d : d;
      if (m != r) {
        float ca = adj[r * NN + m] *
                   (1.f - rw[r * NN + m] * fabsf(rr - s_rsc[m])) * (mr * s_m[m]);
        if (ca == 0.f) d = NEGV;
      }
      s_att[swA(r, m, 4)] = (__bf16)(expf(d - mx) * inv);
    }
  }
  __syncthreads();
  int lane = tid & 31, w = tid >> 5;
  int m0 = w * 16;
  int hi = (lane >> 4) & 1;
  const __bf16* vb = vin + (long long)b * NN * NPADV;  // B-swizzled
  // hoist this wave's 4 A tiles into registers once (32 VGPRs)
  v16bf Ar[4];
#pragma unroll
  for (int kt = 0; kt < 4; ++kt) Ar[kt] = ldTile(s_att, w * 4 + kt, lane);
#pragma unroll 1
  for (int nt = 0; nt < 9; ++nt) {
    int n0 = nt * 16;
    v8f acc;
#pragma unroll
    for (int j = 0; j < 8; ++j) acc[j] = 0.f;
#pragma unroll
    for (int kt = 0; kt < 4; ++kt)
      acc = wmma_bf16(Ar[kt], ldTile(vb, kt * 9 + nt, lane), acc);
    int ncol = n0 + (lane & 15);
    if (ncol < CPAD) {
#pragma unroll
      for (int j = 0; j < 8; ++j)
        catt[((long long)b * NN + m0 + j + hi * 8) * CPAD + ncol] = acc[j];
    }
  }
}

// ---- per step: hypernetwork gates (WMMA per node) + GRU update ----------
__global__ void k_gate(int t, const float* obs, const float* ai, const float* vt,
                       const int* mask, float* h, const float* catt,
                       const __bf16* Wn, const float* bn, const int* lengths,
                       float* out) {
  __shared__ __attribute__((aligned(16))) __bf16 s_X[BB * KPAD];   // 10 KB, A-swizzled
  __shared__ float s_r[BB][DM], s_u[BB][DM], s_hr[BB][DM], s_c[BB][DM];
  int n = blockIdx.x, tid = threadIdx.x;
  int lane = tid & 31, w = tid >> 5;
  int mt = w >> 2, nt = w & 3;
  int m0 = mt * 16, n0 = nt * 16;
  int ocol = n0 + (lane & 15);
  int hi = (lane >> 4) & 1;

  // warm L2 for the candidate-gate weights used later (global_prefetch_b8)
  __builtin_prefetch(Wn + (long long)(2 * NN + n) * KPAD * DM + (long long)tid * 64, 0, 1);

  for (int idx = tid; idx < BB * KPAD; idx += 256) {
    int b = idx / KPAD, c = idx % KPAD;
    float v = (c < CC) ? catt[((long long)b * NN + n) * CPAD + c] : 0.f;
    s_X[swA(b, c, 5)] = (__bf16)v;
  }
  __syncthreads();
  // A tiles shared between the u-gate and r-gate GEMMs: load once
  {
    v16bf Ar[5];
#pragma unroll
    for (int kt = 0; kt < 5; ++kt) Ar[kt] = ldTile(s_X, mt * 5 + kt, lane);
#pragma unroll 1
    for (int g = 0; g < 2; ++g) {  // g=0: update gate, g=1: reset gate
      const __bf16* Wg = Wn + (long long)(g * NN + n) * KPAD * DM;
      float bias = bn[(g * NN + n) * DM + ocol];
      v8f acc;
#pragma unroll
      for (int j = 0; j < 8; ++j) acc[j] = bias;
#pragma unroll
      for (int kt = 0; kt < 5; ++kt)
        acc = wmma_bf16(Ar[kt], ldTile(Wg, kt * 4 + nt, lane), acc);
#pragma unroll
      for (int j = 0; j < 8; ++j) {
        int m = m0 + j + hi * 8;
        float s = 1.f / (1.f + expf(-acc[j]));
        if (g == 0) s_u[m][ocol] = s; else s_r[m][ocol] = s;
      }
    }
  }
  __syncthreads();
  for (int idx = tid; idx < BB * DM; idx += 256) {
    int b = idx / DM, o = idx % DM;
    float hv = h[((long long)b * NN + n) * DM + o];
    float m  = (float)mask[(b * TT + t) * NN + n];
    s_hr[b][o] = (m > 0.f) ? s_r[b][o] * hv : hv;
  }
  __syncthreads();
  for (int idx = tid; idx < BB * KPAD; idx += 256) {
    int b = idx / KPAD, c = idx % KPAD;
    float v;
    if (c < DM)       v = obs[((b * TT + t) * NN + n) * DM + c];
    else if (c == DM) v = 0.5f * tanhf(ai[(b * TT + t) * NN + n] / (vt[b * NN + n] + 1.f));
    else if (c < CC)  v = s_hr[b][c - DM - 1];
    else              v = 0.f;
    s_X[swA(b, c, 5)] = (__bf16)v;
  }
  __syncthreads();
  {
    const __bf16* Wg = Wn + (long long)(2 * NN + n) * KPAD * DM;
    float bias = bn[(2 * NN + n) * DM + ocol];
    v8f acc;
#pragma unroll
    for (int j = 0; j < 8; ++j) acc[j] = bias;
#pragma unroll
    for (int kt = 0; kt < 5; ++kt)
      acc = wmma_bf16(ldTile(s_X, mt * 5 + kt, lane), ldTile(Wg, kt * 4 + nt, lane), acc);
#pragma unroll
    for (int j = 0; j < 8; ++j) s_c[m0 + j + hi * 8][ocol] = tanhf(acc[j]);
  }
  __syncthreads();
  for (int idx = tid; idx < BB * DM; idx += 256) {
    int b = idx / DM, o = idx % DM;
    float hv = h[((long long)b * NN + n) * DM + o];
    float m  = (float)mask[(b * TT + t) * NN + n];
    float u = s_u[b][o], hr = s_hr[b][o], cd = s_c[b][o];
    float nh = (1.f - u) * hr + u * cd;
    float ho = (m > 0.f) ? nh : hv;
    h[((long long)b * NN + n) * DM + o] = ho;
    int ti = lengths[b] - 1;
    if (ti < 0) ti = 0; if (ti > TT - 1) ti = TT - 1;
    if (t == ti) out[((long long)b * NN + n) * DM + o] = ho;
  }
}

extern "C" void kernel_launch(void* const* d_in, const int* in_sizes, int n_in,
                              void* d_out, int out_size, void* d_ws, size_t ws_size,
                              hipStream_t stream) {
  const float* obs  = (const float*)d_in[0];
  const float* ai   = (const float*)d_in[1];
  const float* plm  = (const float*)d_in[2];
  const float* rw   = (const float*)d_in[3];
  const float* pfW1 = (const float*)d_in[4];  const float* pfb1 = (const float*)d_in[5];
  const float* pfW2 = (const float*)d_in[6];  const float* pfb2 = (const float*)d_in[7];
  const float* pgW1 = (const float*)d_in[8];  const float* pgb1 = (const float*)d_in[9];
  const float* pgW2 = (const float*)d_in[10]; const float* pgb2 = (const float*)d_in[11];
  const float* Wq = (const float*)d_in[12];  const float* bq = (const float*)d_in[13];
  const float* Wk = (const float*)d_in[14];  const float* bk = (const float*)d_in[15];
  const float* Wv = (const float*)d_in[16];  const float* bv = (const float*)d_in[17];
  const float* ugW = (const float*)d_in[18]; const float* ugb = (const float*)d_in[19];
  const float* rgW = (const float*)d_in[20]; const float* rgb = (const float*)d_in[21];
  const float* cgW = (const float*)d_in[22]; const float* cgb = (const float*)d_in[23];
  const int* mask    = (const int*)d_in[24];
  const int* lengths = (const int*)d_in[25];
  float* outp = (float*)d_out;

  char* ws = (char*)d_ws;
  size_t off = 0;
  auto alloc = [&](size_t bytes) -> void* {
    void* p = ws + off;
    off = (off + bytes + 255) & ~(size_t)255;
    return p;
  };
  float*  qv   = (float*) alloc((size_t)NN * QD * 4);
  float*  embn = (float*) alloc((size_t)NN * NE * 4);
  float*  adj  = (float*) alloc((size_t)NN * NN * 4);
  float*  vt   = (float*) alloc((size_t)BB * NN * 4);
  float*  bn   = (float*) alloc((size_t)3 * NN * DM * 4);
  float*  hbuf = (float*) alloc((size_t)BB * NN * DM * 4);
  float*  qb   = (float*) alloc((size_t)BB * NN * DK * 4);
  float*  kb   = (float*) alloc((size_t)BB * NN * DK * 4);
  float*  catt = (float*) alloc((size_t)BB * NN * CPAD * 4);
  __bf16* Wn   = (__bf16*)alloc((size_t)3 * NN * KPAD * DM * 2);
  __bf16* Wvb  = (__bf16*)alloc((size_t)KPAD * NPADV * 2);
  __bf16* vb   = (__bf16*)alloc((size_t)BB * NN * NPADV * 2);
  (void)ws_size; (void)in_sizes; (void)n_in; (void)out_size;

  hipMemsetAsync(hbuf, 0, (size_t)BB * NN * DM * 4, stream);

  k_precompute<<<1, 256, 0, stream>>>(plm, pfW1, pfb1, pfW2, pfb2,
                                      pgW1, pgb1, pgW2, pgb2, qv, embn);
  k_adj<<<1, 128, 0, stream>>>(embn, adj);
  k_vartotal<<<16, 256, 0, stream>>>(mask, vt);
  long long totg = 3LL * NN * KPAD * DM + 3 * NN * DM + (long long)KPAD * NPADV;
  k_genw<<<(int)((totg + 255) / 256), 256, 0, stream>>>(qv, ugW, ugb, rgW, rgb,
                                                        cgW, cgb, Wv, Wn, bn, Wvb);
  for (int t = 0; t < TT; ++t) {
    k_qkv<<<BB, 256, 0, stream>>>(t, obs, ai, vt, hbuf, Wq, bq, Wk, bk, bv,
                                  Wvb, qb, kb, vb);
    k_att<<<BB, 256, 0, stream>>>(t, ai, vt, mask, rw, adj, qb, kb, vb, catt);
    k_gate<<<NN, 256, 0, stream>>>(t, obs, ai, vt, mask, hbuf, catt, Wn, bn,
                                   lengths, outp);
  }
}
